// CRaWl_2920577761996
// MI455X (gfx1250) — matrix operand order repaired and checked by reading, not compiled, not run on hardware
//
#include <hip/hip_runtime.h>
#include <hip/hip_bf16.h>
#include <stdint.h>

// ---------------- types ----------------
typedef __bf16 bf16_t;
typedef __attribute__((ext_vector_type(16))) __bf16 v16bf;
typedef __attribute__((ext_vector_type(8)))  __bf16 v8bf;
typedef __attribute__((ext_vector_type(8)))  float  v8f;

#define NNODES 50000
#define NEDGES 200000
#define NWALKS 50000
#define LWALK  17
#define KDW    9
#define POOLO  4
#define LCONV  9        // L - K + 1
#define HIDC   128
#define NGRAPH 256
#define EPSV   1e-5f

#define BM 128
#define BN 128
#define KC 64

// =====================================================================
// WMMA GEMM:  D[M,N] = f(A[M,K]) * B[N,K]^T   (B row-major [N][K], bf16)
// Wave tiling: 8 waves/block; wave = (wm 0..3) x (wn 0..1); each wave owns
// 2 M-tiles (32 rows) x 4 N-tiles (64 cols) so each B fragment feeds two
// back-to-back WMMAs and LDS loads per WMMA drop from 2.25 to 1.5 b128.
// AT: float or bf16. BNRELU_A: a = relu(a*scaleK[k]+biasK[k]) at staging.
// EPI: 0 = store f32, 1 = store bf16, 2 = accumulate into f32,
//      3 = relu then atomicAdd into f32 rows sidx[m] (scatter-mean numer.)
// =====================================================================
template<typename AT, int EPI, bool BNRELU_A>
__global__ __launch_bounds__(256)
void gemm_wmma(const AT* __restrict__ A, const bf16_t* __restrict__ B,
               void* __restrict__ Out, int M, int K, int N,
               const float* __restrict__ scaleK, const float* __restrict__ biasK,
               const int* __restrict__ sidx)
{
    __shared__ alignas(16) bf16_t As[BM * KC];
    __shared__ alignas(16) bf16_t Bs[BN * KC];

    const int m0   = blockIdx.x * BM;
    const int n0   = blockIdx.y * BN;
    const int tid  = threadIdx.x;
    const int wave = tid >> 5;
    const int lane = tid & 31;
    const int wm   = wave >> 1;     // 0..3 : M-tile pair (rows 32*wm .. +31)
    const int wn   = wave & 1;      // 0..1 : N-tile quad (cols 64*wn .. +63)
    const int khalf = lane >> 4;
    const int l15   = lane & 15;
    const int mrow0 = (wm << 5) | l15;   // rows of A fragment 0
    const int mrow1 = mrow0 + 16;        // rows of A fragment 1

    v8f acc[2][4];
    for (int mi = 0; mi < 2; ++mi)
        for (int j = 0; j < 4; ++j)
            for (int e = 0; e < 8; ++e) acc[mi][j][e] = 0.0f;

    for (int kc = 0; kc < K; kc += KC) {
        // ---- stage A (128 rows x 64 k) and B (128 n x 64 k) ----
        {
            const int r     = tid >> 1;
            const int cbase = (tid & 1) * 32;
            const int gm    = m0 + r;
            const int gn    = n0 + r;
            // prefetch next K-chunk of the streamed A matrix into cache
            if (kc + KC < K && gm < M)
                __builtin_prefetch(&A[(size_t)gm * K + kc + KC + cbase], 0, 0);
            #pragma unroll 8
            for (int c = 0; c < 32; ++c) {
                const int k = kc + cbase + c;
                float v = 0.0f;
                if (gm < M && k < K) {
                    v = (float)A[(size_t)gm * K + k];
                    if (BNRELU_A) {
                        v = v * scaleK[k] + biasK[k];
                        v = v > 0.0f ? v : 0.0f;
                    }
                }
                As[r * KC + cbase + c] = (bf16_t)v;
                bf16_t bv = (bf16_t)0.0f;
                if (k < K) bv = B[(size_t)gn * K + k];
                Bs[r * KC + cbase + c] = bv;
            }
        }
        __syncthreads();

        #pragma unroll
        for (int ks = 0; ks < KC; ks += 32) {
            // A fragments: per lane, elements j=0..7 -> K=khalf*8+j,
            //              j=8..15 -> K=16+khalf*8+(j-8)
            const bf16_t* ar0 = &As[mrow0 * KC + ks + khalf * 8];
            v8bf a0lo = *(const v8bf*)(ar0);
            v8bf a0hi = *(const v8bf*)(ar0 + 16);
            v16bf a0 = __builtin_shufflevector(a0lo, a0hi,
                       0,1,2,3,4,5,6,7,8,9,10,11,12,13,14,15);
            const bf16_t* ar1 = &As[mrow1 * KC + ks + khalf * 8];
            v8bf a1lo = *(const v8bf*)(ar1);
            v8bf a1hi = *(const v8bf*)(ar1 + 16);
            v16bf a1 = __builtin_shufflevector(a1lo, a1hi,
                       0,1,2,3,4,5,6,7,8,9,10,11,12,13,14,15);
            #pragma unroll
            for (int j = 0; j < 4; ++j) {
                // B fragment: N = lane&15 within tile, K same as A (khalf*8+j, 16+khalf*8+(j-8))
                const int ntile = (wn << 2) + j;
                const bf16_t* br = &Bs[((ntile << 4) | l15) * KC + ks + khalf * 8];
                v8bf blo = *(const v8bf*)(br);
                v8bf bhi = *(const v8bf*)(br + 16);
                v16bf b = __builtin_shufflevector(blo, bhi,
                          0,1,2,3,4,5,6,7,8,9,10,11,12,13,14,15);
                // two WMMAs share this B fragment -> issue back-to-back
                acc[0][j] = __builtin_amdgcn_wmma_f32_16x16x32_bf16(
                    false, a0, false, b, (short)0, acc[0][j], false, false);
                acc[1][j] = __builtin_amdgcn_wmma_f32_16x16x32_bf16(
                    false, a1, false, b, (short)0, acc[1][j], false, false);
            }
        }
        __syncthreads();
    }

    // ---- epilogue: D layout -> M = r + khalf*8, N = lane&15 ----
    for (int mi = 0; mi < 2; ++mi) {
        const int mbase = m0 + (wm << 5) + (mi << 4) + khalf * 8;
        for (int j = 0; j < 4; ++j) {
            const int n = n0 + (((wn << 2) + j) << 4) + l15;
            for (int r = 0; r < 8; ++r) {
                const int m = mbase + r;
                if (m >= M) continue;
                float v = acc[mi][j][r];
                if (EPI == 0) {
                    ((float*)Out)[(size_t)m * N + n] = v;
                } else if (EPI == 1) {
                    ((bf16_t*)Out)[(size_t)m * N + n] = (bf16_t)v;
                } else if (EPI == 2) {
                    float* o = (float*)Out + (size_t)m * N + n;
                    *o = *o + v;
                } else {                      // EPI == 3
                    v = v > 0.0f ? v : 0.0f;
                    atomicAdd((float*)Out + (size_t)sidx[m] * N + n, v);
                }
            }
        }
    }
}

// ------------- gather + concat: Xin[w*L+l, :] (bf16) -------------
__global__ void gather_xin(bf16_t* __restrict__ Xin,
                           const float* __restrict__ h, int din,
                           const float* __restrict__ edge_attr,
                           const float* __restrict__ walk_x,
                           const int* __restrict__ walk_nodes,
                           const int* __restrict__ walk_edges,
                           int cin, size_t total)
{
    size_t i = (size_t)blockIdx.x * blockDim.x + threadIdx.x;
    if (i >= total) return;
    const int k = (int)(i % cin);
    const size_t m = i / cin;
    const int w = (int)(m / LWALK);
    const int l = (int)(m % LWALK);
    float v;
    if (k < din) {
        v = h[(size_t)walk_nodes[m] * din + k];
    } else if (k < din + 16) {
        v = (l == 0) ? 0.0f
            : edge_attr[(size_t)walk_edges[(size_t)w * (LWALK - 1) + (l - 1)] * 16 + (k - din)];
    } else {
        v = walk_x[((size_t)w * 16 + (k - din - 16)) * LWALK + l];
    }
    Xin[i] = (bf16_t)v;
}

// ------------- depthwise conv K=9 (valid): Y1[W,L,128] -> Yd[W,9,128] ---
__global__ void depthwise_k9(const bf16_t* __restrict__ Y1,
                             const float* __restrict__ W2, // [128][9]
                             bf16_t* __restrict__ Yd)
{
    const int w = blockIdx.x;
    const int c = threadIdx.x;            // 128 threads
    float y[LWALK];
    for (int l = 0; l < LWALK; ++l)
        y[l] = (float)Y1[((size_t)w * LWALK + l) * 128 + c];
    float k9[KDW];
    for (int j = 0; j < KDW; ++j) k9[j] = W2[c * KDW + j];
    for (int t = 0; t < LCONV; ++t) {
        float s = 0.0f;
        #pragma unroll
        for (int j = 0; j < KDW; ++j) s += k9[j] * y[t + j];
        Yd[((size_t)w * LCONV + t) * 128 + c] = (bf16_t)s;
    }
}

// ------------- per-column sum/sumsq (blockDim.x == C) -------------
template<typename T>
__global__ void col_stats(const T* __restrict__ X, int M, int C,
                          float* __restrict__ sums, float* __restrict__ sumsq)
{
    const int c = threadIdx.x;
    const int r0 = blockIdx.x * 64;
    int rend = r0 + 64; if (rend > M) rend = M;
    float s = 0.0f, s2 = 0.0f;
    for (int r = r0; r < rend; ++r) {
        float v = (float)X[(size_t)r * C + c];
        s += v; s2 += v * v;
    }
    atomicAdd(&sums[c], s);
    atomicAdd(&sumsq[c], s2);
}

__global__ void bn_finalize(const float* sums, const float* sumsq,
                            const float* g, const float* b,
                            float* scale, float* bias, int C, float invM)
{
    const int c = threadIdx.x;
    if (c < C) {
        float mean = sums[c] * invM;
        float var  = sumsq[c] * invM - mean * mean;
        float sc   = g[c] * rsqrtf(var + EPSV);
        scale[c] = sc;
        bias[c]  = b[c] - mean * sc;
    }
}

// ------------- misc elementwise / scatter kernels -------------
__global__ void idx_flat_k(int* __restrict__ IDX, const int* __restrict__ walk_nodes)
{
    size_t i = (size_t)blockIdx.x * blockDim.x + threadIdx.x;
    if (i >= (size_t)NWALKS * LCONV) return;
    const int w = (int)(i / LCONV);
    const int t = (int)(i % LCONV);
    IDX[i] = walk_nodes[(size_t)w * LWALK + POOLO + t];
}

__global__ void counts_k(float* __restrict__ CNT, const int* __restrict__ IDX, size_t n)
{
    size_t i = (size_t)blockIdx.x * blockDim.x + threadIdx.x;
    if (i < n) atomicAdd(&CNT[IDX[i]], 1.0f);
}

__global__ void pn_div_k(float* __restrict__ PN, const float* __restrict__ CNT, size_t n)
{
    size_t i = (size_t)blockIdx.x * blockDim.x + threadIdx.x;
    if (i < n) PN[i] /= fmaxf(CNT[i >> 7], 1.0f);
}

__global__ void seg_sum_k(const float* __restrict__ H, const int* __restrict__ batch,
                          float* __restrict__ G, size_t n)
{
    size_t i = (size_t)blockIdx.x * blockDim.x + threadIdx.x;
    if (i < n) atomicAdd(&G[(size_t)batch[i >> 7] * 128 + (i & 127)], H[i]);
}

__global__ void add_vec_k(float* __restrict__ A, const float* __restrict__ B, size_t n)
{
    size_t i = (size_t)blockIdx.x * blockDim.x + threadIdx.x;
    if (i < n) A[i] += B[i];
}

__global__ void h_add_vnh_k(float* __restrict__ H, const float* __restrict__ VNH,
                            const int* __restrict__ batch, size_t n)
{
    size_t i = (size_t)blockIdx.x * blockDim.x + threadIdx.x;
    if (i < n) H[i] += VNH[(size_t)batch[i >> 7] * 128 + (i & 127)];
}

__global__ void final_relu_k(const float* __restrict__ H, const float* __restrict__ scale,
                             const float* __restrict__ bias, float* __restrict__ out, size_t n)
{
    size_t i = (size_t)blockIdx.x * blockDim.x + threadIdx.x;
    if (i < n) {
        const int c = (int)(i & 127);
        out[i] = fmaxf(H[i] * scale[c] + bias[c], 0.0f);
    }
}

// small f32 matmul for the virtual-node path: Out[m,n] = f(A[m,:]) . W[:,n]
__global__ void small_mm_k(const float* __restrict__ A, const float* __restrict__ Wkn,
                           float* __restrict__ Out, int K, int N, int bnrelu,
                           const float* __restrict__ scale, const float* __restrict__ bias)
{
    const int m = blockIdx.x;
    const int n = threadIdx.x;
    float s = 0.0f;
    for (int k = 0; k < K; ++k) {
        float a = A[(size_t)m * K + k];
        if (bnrelu) { a = fmaxf(a * scale[k] + bias[k], 0.0f); }
        s += a * Wkn[(size_t)k * N + n];
    }
    Out[(size_t)m * N + n] = s;
}

// weight prep: f32 [R,C] -> bf16, optional transpose to [C][R]
__global__ void cvt_weight_k(bf16_t* __restrict__ dst, const float* __restrict__ src,
                             int R, int C, int transpose)
{
    size_t i = (size_t)blockIdx.x * blockDim.x + threadIdx.x;
    if (i >= (size_t)R * C) return;
    const int r = (int)(i / C), c = (int)(i % C);
    if (transpose) dst[(size_t)c * R + r] = (bf16_t)src[i];
    else           dst[i] = (bf16_t)src[i];
}

// =====================================================================
extern "C" void kernel_launch(void* const* d_in, const int* in_sizes, int n_in,
                              void* d_out, int out_size, void* d_ws, size_t ws_size,
                              hipStream_t stream)
{
    (void)in_sizes; (void)n_in; (void)out_size; (void)ws_size;

    // --- input mapping: top-level insertion order; params in pytree (sorted-key) order
    const float* x         = (const float*)d_in[0];
    const float* edge_attr = (const float*)d_in[1];
    const float* walk_x    = (const float*)d_in[2];
    const float* fin_b     = (const float*)d_in[3];
    const float* fin_g     = (const float*)d_in[4];

    struct LayerP { const float *bn1_b,*bn1_g,*conv1,*conv2,*conv3,*obn_b,*obn_g,*out1,*out2,*rescale; int cin; };
    LayerP LP[3];
    int base = 5;
    for (int i = 0; i < 3; ++i) {
        LP[i].bn1_b = (const float*)d_in[base + 0];
        LP[i].bn1_g = (const float*)d_in[base + 1];
        LP[i].conv1 = (const float*)d_in[base + 2];
        LP[i].conv2 = (const float*)d_in[base + 3];
        LP[i].conv3 = (const float*)d_in[base + 4];
        LP[i].obn_b = (const float*)d_in[base + 5];
        LP[i].obn_g = (const float*)d_in[base + 6];
        LP[i].out1  = (const float*)d_in[base + 7];
        LP[i].out2  = (const float*)d_in[base + 8];
        LP[i].rescale = (i == 0) ? (const float*)d_in[base + 9] : nullptr;
        LP[i].cin = (i == 0) ? 64 : 160;
        base += (i == 0) ? 10 : 9;
    }
    struct VnP { const float *bn_b,*bn_g,*lin1,*lin2; } VP[2];
    for (int v = 0; v < 2; ++v) {
        VP[v].bn_b = (const float*)d_in[base + 0];
        VP[v].bn_g = (const float*)d_in[base + 1];
        VP[v].lin1 = (const float*)d_in[base + 2];
        VP[v].lin2 = (const float*)d_in[base + 3];
        base += 4;
    }
    const int* walk_nodes = (const int*)d_in[base + 0];
    const int* walk_edges = (const int*)d_in[base + 1];
    const int* batch      = (const int*)d_in[base + 2];

    // --- workspace layout
    char* wsb = (char*)d_ws;
    size_t off = 0;
    auto alloc = [&](size_t bytes) -> void* {
        void* p = wsb + off;
        off += (bytes + 255) & ~(size_t)255;
        return p;
    };
    float* H2    = (float*)alloc((size_t)NNODES * 128 * 4);
    float* PN    = (float*)alloc((size_t)NNODES * 128 * 4);
    float* CNT   = (float*)alloc((size_t)NNODES * 4);
    int*   IDX   = (int*)  alloc((size_t)NWALKS * LCONV * 4);
    float* STATS = (float*)alloc(4 * 256 * 4);
    float* sums  = STATS, *sumsq = STATS + 256, *scale = STATS + 512, *bias = STATS + 768;
    float* G     = (float*)alloc(256 * 128 * 4);
    float* T1    = (float*)alloc(256 * 128 * 4);
    float* VNH   = (float*)alloc(256 * 128 * 4);
    bf16_t* conv1bf[3], *conv3bf[3], *out1T[3], *out2T[3];
    for (int i = 0; i < 3; ++i) {
        conv1bf[i] = (bf16_t*)alloc((size_t)128 * LP[i].cin * 2);
        conv3bf[i] = (bf16_t*)alloc((size_t)128 * 128 * 2);
        out1T[i]   = (bf16_t*)alloc((size_t)256 * 128 * 2);
        out2T[i]   = (bf16_t*)alloc((size_t)128 * 256 * 2);
    }
    bf16_t* rescT = (bf16_t*)alloc((size_t)128 * 32 * 2);
    // big regions (aliased across pipeline phases)
    bf16_t* R1 = (bf16_t*)alloc((size_t)NWALKS * LWALK * 160 * 2); // XIN, later YD
    bf16_t* R2 = (bf16_t*)alloc((size_t)NWALKS * LWALK * 128 * 2); // Y1, later Q
    bf16_t* XIN = R1;
    bf16_t* YD  = R1;
    bf16_t* Y1  = R2;
    float*  Q   = (float*)R2;

    const int MW   = NWALKS * LWALK;   // 850000
    const int MD   = NWALKS * LCONV;   // 450000
    const size_t NH = (size_t)NNODES * 128;

    auto blocks = [](size_t n) { return (unsigned)((n + 255) / 256); };

    // --- weight prep (f32 -> bf16, [O][K] layout) ---
    for (int i = 0; i < 3; ++i) {
        cvt_weight_k<<<blocks((size_t)128 * LP[i].cin), 256, 0, stream>>>(conv1bf[i], LP[i].conv1, 128, LP[i].cin, 0);
        cvt_weight_k<<<blocks(128 * 128), 256, 0, stream>>>(conv3bf[i], LP[i].conv3, 128, 128, 0);
        cvt_weight_k<<<blocks(128 * 256), 256, 0, stream>>>(out1T[i], LP[i].out1, 128, 256, 1);
        cvt_weight_k<<<blocks(256 * 128), 256, 0, stream>>>(out2T[i], LP[i].out2, 256, 128, 1);
    }
    cvt_weight_k<<<blocks(32 * 128), 256, 0, stream>>>(rescT, LP[0].rescale, 32, 128, 1);

    // --- scatter indices + counts (layer-invariant) ---
    idx_flat_k<<<blocks(MD), 256, 0, stream>>>(IDX, walk_nodes);
    hipMemsetAsync(CNT, 0, (size_t)NNODES * 4, stream);
    counts_k<<<blocks(MD), 256, 0, stream>>>(CNT, IDX, (size_t)MD);

    const float* hcur = x;
    int din = 32;

    for (int i = 0; i < 3; ++i) {
        const int cin = LP[i].cin;
        // 1) gather/concat -> XIN (bf16)
        gather_xin<<<blocks((size_t)MW * cin), 256, 0, stream>>>(
            XIN, hcur, din, edge_attr, walk_x, walk_nodes, walk_edges, cin, (size_t)MW * cin);
        // 2) conv1: Y1 = XIN * conv1^T (bf16 out)
        gemm_wmma<bf16_t, 1, false><<<dim3((MW + BM - 1) / BM, 1), 256, 0, stream>>>(
            XIN, conv1bf[i], Y1, MW, cin, 128, nullptr, nullptr, nullptr);
        // 3) depthwise K=9 -> YD (bf16)
        depthwise_k9<<<NWALKS, 128, 0, stream>>>(Y1, LP[i].conv2, YD);
        // 4) bn1 batch-stats over [MD,128]
        hipMemsetAsync(STATS, 0, 2 * 256 * 4, stream);
        col_stats<bf16_t><<<(MD + 63) / 64, 128, 0, stream>>>(YD, MD, 128, sums, sumsq);
        bn_finalize<<<1, 128, 0, stream>>>(sums, sumsq, LP[i].bn1_g, LP[i].bn1_b,
                                           scale, bias, 128, 1.0f / MD);
        // 5) conv3 with fused bn+relu on A, relu + scatter-add into PN
        hipMemsetAsync(PN, 0, NH * 4, stream);
        gemm_wmma<bf16_t, 3, true><<<dim3((MD + BM - 1) / BM, 1), 256, 0, stream>>>(
            YD, conv3bf[i], PN, MD, 128, 128, scale, bias, IDX);
        pn_div_k<<<blocks(NH), 256, 0, stream>>>(PN, CNT, NH);
        // 6) rescale (layer 0): H2 = x @ rescale
        if (i == 0)
            gemm_wmma<float, 0, false><<<dim3((NNODES + BM - 1) / BM, 1), 256, 0, stream>>>(
                x, rescT, H2, NNODES, 32, 128, nullptr, nullptr, nullptr);
        // 7) Q = PN @ out1  (f32 out, bn comes after)
        gemm_wmma<float, 0, false><<<dim3((NNODES + BM - 1) / BM, 2), 256, 0, stream>>>(
            PN, out1T[i], Q, NNODES, 128, 256, nullptr, nullptr, nullptr);
        // 8) obn stats over [N,256]
        hipMemsetAsync(STATS, 0, 2 * 256 * 4, stream);
        col_stats<float><<<(NNODES + 63) / 64, 256, 0, stream>>>(Q, NNODES, 256, sums, sumsq);
        bn_finalize<<<1, 256, 0, stream>>>(sums, sumsq, LP[i].obn_g, LP[i].obn_b,
                                           scale, bias, 256, 1.0f / NNODES);
        // 9) H2 += relu(bn(Q)) @ out2
        gemm_wmma<float, 2, true><<<dim3((NNODES + BM - 1) / BM, 1), 256, 0, stream>>>(
            Q, out2T[i], H2, NNODES, 256, 128, scale, bias, nullptr);
        hcur = H2; din = 128;
        // 10) virtual node
        if (i < 2) {
            hipMemsetAsync(G, 0, 256 * 128 * 4, stream);
            seg_sum_k<<<blocks(NH), 256, 0, stream>>>(H2, batch, G, NH);
            if (i > 0) add_vec_k<<<blocks(256 * 128), 256, 0, stream>>>(G, VNH, 256 * 128);
            small_mm_k<<<256, 128, 0, stream>>>(G, VP[i].lin1, T1, 128, 128, 0, nullptr, nullptr);
            hipMemsetAsync(STATS, 0, 2 * 256 * 4, stream);
            col_stats<float><<<(256 + 63) / 64, 128, 0, stream>>>(T1, 256, 128, sums, sumsq);
            bn_finalize<<<1, 128, 0, stream>>>(sums, sumsq, VP[i].bn_g, VP[i].bn_b,
                                               scale, bias, 128, 1.0f / 256.0f);
            small_mm_k<<<256, 128, 0, stream>>>(T1, VP[i].lin2, VNH, 128, 128, 1, scale, bias);
            h_add_vnh_k<<<blocks(NH), 256, 0, stream>>>(H2, VNH, batch, NH);
        }
    }

    // final bn + relu -> d_out
    hipMemsetAsync(STATS, 0, 2 * 256 * 4, stream);
    col_stats<float><<<(NNODES + 63) / 64, 128, 0, stream>>>(H2, NNODES, 128, sums, sumsq);
    bn_finalize<<<1, 128, 0, stream>>>(sums, sumsq, fin_g, fin_b, scale, bias, 128, 1.0f / NNODES);
    final_relu_k<<<blocks(NH), 256, 0, stream>>>(H2, scale, bias, (float*)d_out, NH);
}
